// ParallelHybridReasoningLayer_30313879175675
// MI455X (gfx1250) — compile-verified, matrix operand order checked by reading, not run on hardware
//
#include <hip/hip_runtime.h>

#define DEV __device__ __forceinline__

typedef __attribute__((ext_vector_type(16))) __bf16 v16bf;
typedef __attribute__((ext_vector_type(8)))  float  v8f;
typedef unsigned int u32x4 __attribute__((ext_vector_type(4)));
typedef int          i32x8 __attribute__((ext_vector_type(8)));
typedef int          i32x4 __attribute__((ext_vector_type(4)));

constexpr int Dm   = 256;
constexpr int Tlen = 2048;
constexpr int Bb   = 4;
constexpr int NTOK = Bb * Tlen; // 8192 token rows

union AFrag { unsigned short s[16]; v16bf v; };

DEV unsigned short f2bf(float f) {
  unsigned int u = __float_as_uint(f);
  unsigned int r = (u + 0x7FFFu + ((u >> 16) & 1u)) >> 16; // RNE
  return (unsigned short)r;
}

DEV v8f wmma_bf16(const AFrag& a, const AFrag& b, v8f c) {
  return __builtin_amdgcn_wmma_f32_16x16x32_bf16(false, a.v, false, b.v,
                                                 (short)0, c, false, false);
}

DEV float act_apply(float v, int act) {
  if (act == 1) return 1.0f / (1.0f + __expf(-v));                   // sigmoid
  if (act == 2) return 0.5f * v * (1.0f + erff(v * 0.70710678118f)); // exact gelu
  if (act == 3) return tanhf(v);
  return v;
}

// f32 -> bf16 weight conversion (one-time)
__global__ void tobf16_kernel(const float* __restrict__ src,
                              unsigned short* __restrict__ dst, int n)
{
  int i = blockIdx.x * 256 + threadIdx.x;
  if (i < n) dst[i] = f2bf(src[i]);
}

// ---------------------------------------------------------------------------
// bf16 WMMA GEMM: C[row, col] = act( sum_k A[row,k]*W[col, wOff+k] (+Cprev)
//                                    (+bias[col]) ) * out_scale (+ resid)
// Tile: 128(M) x 64(N) per workgroup, BK=32, 8 waves, each wave 16x64.
// Weight tile (64x32 bf16) is DMA'd into LDS by the Tensor Data Mover.
// ---------------------------------------------------------------------------
__global__ __launch_bounds__(256) void gemm_bf16_kernel(
    const float* __restrict__ A, int lda,
    const unsigned short* __restrict__ Wbf, int ldw, int wOff,
    const float* __restrict__ bias,
    float* __restrict__ C, int ldc,
    const float* __restrict__ resid,
    int K, int accumulate, int act, float out_scale)
{
  __shared__ unsigned short As[128][32]; // 8 KB
  __shared__ unsigned short Ws[64][32];  // 4 KB (TDM destination)

  const int tid    = threadIdx.x;
  const int wave   = tid >> 5;
  const int lane   = tid & 31;
  const int lanelo = lane & 15;
  const int hi     = lane >> 4;
  const int m0     = blockIdx.x * 128;
  const int n0     = blockIdx.y * 64;
  const int kb0    = hi * 8;

  v8f acc[4];
  for (int nt = 0; nt < 4; ++nt)
    for (int r = 0; r < 8; ++r) acc[nt][r] = 0.0f;

  const int nKB = K >> 5;
  for (int kb = 0; kb < nKB; ++kb) {
    const int k0 = kb << 5;

#if __has_builtin(__builtin_amdgcn_tensor_load_to_lds)
    if (wave == 0) {
      // D# for a 2D bf16 tile: 64 rows x 32 elems, row stride = ldw elems.
      unsigned long long gaddr =
          (unsigned long long)(const void*)(Wbf + (size_t)n0 * ldw + wOff + k0);
      unsigned lds_off = (unsigned)(unsigned long long)(const void*)&Ws[0][0];
      u32x4 g0;
      g0[0] = 1u;                                            // count=1 (valid user D#)
      g0[1] = lds_off;                                       // lds_addr (bytes)
      g0[2] = (unsigned)(gaddr & 0xFFFFFFFFull);             // global_addr[31:0]
      g0[3] = (unsigned)((gaddr >> 32) & 0x1FFFFFFull)       // global_addr[56:32]
              | (2u << 30);                                  // type = 2 ("image")
      i32x8 g1;
      g1[0] = (int)(1u << 16);   // data_size=1 (2 bytes), no multicast/pad/iterate
      g1[1] = (int)(32u << 16);  // tensor_dim0 = 32 (low half at bits 79:48)
      g1[2] = (int)(64u << 16);  // tensor_dim0 hi=0 | tensor_dim1 = 64 (low half)
      g1[3] = (int)(32u << 16);  // tensor_dim1 hi=0 | tile_dim0 = 32
      g1[4] = (int)64u;          // tile_dim1 = 64 | tile_dim2 = 0
      g1[5] = (int)(unsigned)ldw;// tensor_dim0_stride[31:0]
      g1[6] = 0;                 // stride hi | tensor_dim1_stride lo
      g1[7] = 0;
      i32x4 gz; gz[0] = gz[1] = gz[2] = gz[3] = 0;
      i32x8 gz8;
      for (int z = 0; z < 8; ++z) gz8[z] = 0;
      __builtin_amdgcn_tensor_load_to_lds(g0, g1, gz, gz, gz8, 0);
    }
#else
    for (int i = tid; i < (64 * 32) / 8; i += 256) {
      int r = i >> 2, c8 = (i & 3) * 8;
      *reinterpret_cast<uint4*>(&Ws[r][c8]) =
          *reinterpret_cast<const uint4*>(&Wbf[(size_t)(n0 + r) * ldw + wOff + k0 + c8]);
    }
#endif

    // Stage A tile (f32 -> bf16), vectorized: float4 load, 8B LDS store.
    for (int i = tid; i < (128 * 32) / 4; i += 256) {
      int r = i >> 3, c4 = (i & 7) * 4;
      const float4 v =
          *reinterpret_cast<const float4*>(&A[(size_t)(m0 + r) * lda + k0 + c4]);
      union { unsigned short us[4]; unsigned long long u; } p;
      p.us[0] = f2bf(v.x); p.us[1] = f2bf(v.y);
      p.us[2] = f2bf(v.z); p.us[3] = f2bf(v.w);
      *reinterpret_cast<unsigned long long*>(&As[r][c4]) = p.u;
    }

    if (kb + 1 < nKB) // prefetch next A K-tile (global_prefetch_b8)
      __builtin_prefetch(&A[(size_t)(m0 + (tid >> 1)) * lda + k0 + 32], 0, 1);

#if __has_builtin(__builtin_amdgcn_tensor_load_to_lds)
    if (wave == 0) __builtin_amdgcn_s_wait_tensorcnt(0);
#endif
    __syncthreads();

    // A fragment: lanes 0-15 & 16-31 both carry rows M=0..15; K halves split by hi.
    AFrag af;
    {
      int r = wave * 16 + lanelo;
      for (int j = 0; j < 8; ++j) {
        af.s[j]     = As[r][kb0 + j];
        af.s[8 + j] = As[r][16 + kb0 + j];
      }
    }
    for (int nt = 0; nt < 4; ++nt) {
      AFrag bf;
      int r = nt * 16 + lanelo;
      for (int j = 0; j < 8; ++j) {
        bf.s[j]     = Ws[r][kb0 + j];
        bf.s[8 + j] = Ws[r][16 + kb0 + j];
      }
      acc[nt] = wmma_bf16(af, bf, acc[nt]);
    }
    __syncthreads();
  }

  // C layout: lane<16 -> M=r, N=lane; lane>=16 -> M=8+r, N=lane-16
  for (int nt = 0; nt < 4; ++nt) {
    for (int r = 0; r < 8; ++r) {
      int row = m0 + wave * 16 + r + 8 * hi;
      int col = n0 + nt * 16 + lanelo;
      size_t idx = (size_t)row * ldc + col;
      float v = acc[nt][r];
      if (accumulate) v += C[idx];
      if (bias) v += bias[col];
      v = act_apply(v, act);
      v *= out_scale;
      if (resid) v += resid[idx];
      C[idx] = v;
    }
  }
}

// ---------------------------------------------------------------------------
// Banded causal attention (WBAND=64, HD=64): one wave per (b, h, 16-query tile)
// ---------------------------------------------------------------------------
__global__ __launch_bounds__(32) void attn_kernel(const float* __restrict__ qkv,
                                                  float* __restrict__ out)
{
  const int qt = blockIdx.x, h = blockIdx.y, b = blockIdx.z;
  const int lane = threadIdx.x, lanelo = lane & 15, hi = lane >> 4;
  const int kb0 = hi * 8;
  const int qb = qt * 16;

  __shared__ float          S[16][96];
  __shared__ unsigned short P[16][96];

  const float* qbase = qkv + ((size_t)(b * Tlen + qb + lanelo)) * 768 + h * 64;
  AFrag qf[2];
  for (int c = 0; c < 2; ++c)
    for (int j = 0; j < 8; ++j) {
      qf[c].s[j]     = f2bf(qbase[c * 32 + kb0 + j]);
      qf[c].s[8 + j] = f2bf(qbase[c * 32 + 16 + kb0 + j]);
    }

  int kt0 = qt - 4; if (kt0 < 0) kt0 = 0;
  const int ntile = qt - kt0 + 1;

  for (int kt = kt0; kt <= qt; ++kt) {
    const float* kbase = qkv + ((size_t)(b * Tlen + kt * 16 + lanelo)) * 768 + 256 + h * 64;
    AFrag kf[2];
    for (int c = 0; c < 2; ++c)
      for (int j = 0; j < 8; ++j) {
        kf[c].s[j]     = f2bf(kbase[c * 32 + kb0 + j]);
        kf[c].s[8 + j] = f2bf(kbase[c * 32 + 16 + kb0 + j]);
      }
    v8f sc;
    for (int r = 0; r < 8; ++r) sc[r] = 0.0f;
    sc = wmma_bf16(qf[0], kf[0], sc);
    sc = wmma_bf16(qf[1], kf[1], sc);
    for (int r = 0; r < 8; ++r) {
      int m = r + 8 * hi;
      int i = qb + m;
      int j = kt * 16 + lanelo;
      float val = (j <= i && j > i - 64) ? sc[r] * 0.125f : -1.0e9f;
      S[m][(kt - kt0) * 16 + lanelo] = val;
    }
  }
  __syncthreads();

  const int w = ntile * 16;
  const int nch = (w + 31) >> 5;
  if (lane < 16) {
    float mx = -1.0e30f;
    for (int c = 0; c < w; ++c) mx = fmaxf(mx, S[lane][c]);
    float sum = 0.0f;
    for (int c = 0; c < w; ++c) sum += __expf(S[lane][c] - mx);
    float inv = 1.0f / sum;
    for (int c = 0; c < w; ++c) P[lane][c] = f2bf(__expf(S[lane][c] - mx) * inv);
    for (int c = w; c < nch * 32; ++c) P[lane][c] = 0;
  }
  __syncthreads();

  v8f acc[4];
  for (int nt = 0; nt < 4; ++nt)
    for (int r = 0; r < 8; ++r) acc[nt][r] = 0.0f;

  for (int c = 0; c < nch; ++c) {
    AFrag pa;
    for (int j = 0; j < 8; ++j) {
      pa.s[j]     = P[lanelo][c * 32 + kb0 + j];
      pa.s[8 + j] = P[lanelo][c * 32 + 16 + kb0 + j];
    }
    for (int nt = 0; nt < 4; ++nt) {
      AFrag bv;
      int dcol = h * 64 + nt * 16 + lanelo;
      for (int j = 0; j < 8; ++j) {
        int k1 = kt0 * 16 + c * 32 + kb0 + j;
        int k2 = kt0 * 16 + c * 32 + 16 + kb0 + j;
        if (k1 > Tlen - 1) k1 = Tlen - 1;   // padded keys have P==0
        if (k2 > Tlen - 1) k2 = Tlen - 1;
        bv.s[j]     = f2bf(qkv[((size_t)(b * Tlen + k1)) * 768 + 512 + dcol]);
        bv.s[8 + j] = f2bf(qkv[((size_t)(b * Tlen + k2)) * 768 + 512 + dcol]);
      }
      acc[nt] = wmma_bf16(pa, bv, acc[nt]);
    }
  }

  for (int nt = 0; nt < 4; ++nt)
    for (int r = 0; r < 8; ++r) {
      int row = b * Tlen + qb + r + 8 * hi;
      out[(size_t)row * Dm + h * 64 + nt * 16 + lanelo] = acc[nt][r];
    }
}

// ---------------------------------------------------------------------------
// Depthwise causal conv, KLOC=8
// ---------------------------------------------------------------------------
__global__ void dwconv8_kernel(const float* __restrict__ x,
                               const float* __restrict__ w,
                               const float* __restrict__ bias,
                               float* __restrict__ y)
{
  int idx = blockIdx.x * 256 + threadIdx.x;
  int d = idx & 255;
  int n = idx >> 8;
  int t = n & (Tlen - 1);
  float s = bias[d];
  for (int tau = 0; tau < 8; ++tau) {
    int tt = t - 7 + tau;
    if (tt >= 0) s += x[(size_t)(n - 7 + tau) * Dm + d] * w[d * 8 + tau];
  }
  y[idx] = s;
}

// Per-(b,d) first-order recurrence: y[t] = decay*y[t-1] + inscale*x[t]
__global__ void scan_kernel(const float* __restrict__ x, float* __restrict__ y,
                            const float* __restrict__ init,
                            float decay, float inscale)
{
  int id = blockIdx.x * 256 + threadIdx.x;
  int b = id >> 8, d = id & 255;
  float s = init ? init[id] : 0.0f;
  const float* xp = x + (size_t)b * Tlen * Dm + d;
  float* yp = y + (size_t)b * Tlen * Dm + d;
  for (int t = 0; t < Tlen; ++t) {
    s = decay * s + inscale * xp[(size_t)t * Dm];
    yp[(size_t)t * Dm] = s;
  }
}

// concepts = cumsum(gk*gv) / (cumsum(gk)+1e-5), gk = elu(.)+1
__global__ void concepts_kernel(const float* __restrict__ gkv,
                                float* __restrict__ out)
{
  int id = blockIdx.x * 256 + threadIdx.x;
  int b = id >> 8, d = id & 255;
  float num = 0.0f, den = 0.0f;
  const float* p = gkv + (size_t)b * Tlen * 512;
  for (int t = 0; t < Tlen; ++t) {
    float k0 = p[(size_t)t * 512 + d];
    float gk = (k0 > 0.0f) ? (k0 + 1.0f) : __expf(k0);
    float gv = p[(size_t)t * 512 + 256 + d];
    num += gk * gv;
    den += gk;
    out[(size_t)(b * Tlen + t) * Dm + d] = num / (den + 1e-5f);
  }
}

__global__ void mul_kernel(const float* __restrict__ a,
                           const float* __restrict__ b,
                           float* __restrict__ o)
{
  int idx = blockIdx.x * 256 + threadIdx.x;
  o[idx] = a[idx] * b[idx];
}

// LayerNorm over D=256: one wave32 per row, 8 elems/lane, shuffle reductions
__global__ __launch_bounds__(256) void ln_kernel(const float* __restrict__ xin,
                                                 const float* __restrict__ resid,
                                                 const float* __restrict__ g,
                                                 const float* __restrict__ bsh,
                                                 float* __restrict__ out)
{
  int wave = threadIdx.x >> 5, lane = threadIdx.x & 31;
  int row = blockIdx.x * 8 + wave;
  size_t base = (size_t)row * Dm;
  float v[8];
  float s = 0.0f;
  for (int j = 0; j < 8; ++j) {
    float t = xin[base + lane + 32 * j];
    if (resid) t += resid[base + lane + 32 * j];
    v[j] = t;
    s += t;
  }
  for (int off = 16; off > 0; off >>= 1) s += __shfl_xor(s, off);
  float mean = s * (1.0f / 256.0f);
  float s2 = 0.0f;
  for (int j = 0; j < 8; ++j) { float d = v[j] - mean; s2 += d * d; }
  for (int off = 16; off > 0; off >>= 1) s2 += __shfl_xor(s2, off);
  float rs = rsqrtf(s2 * (1.0f / 256.0f) + 1e-5f);
  for (int j = 0; j < 8; ++j) {
    int c = lane + 32 * j;
    out[base + c] = (v[j] - mean) * rs * g[c] + bsh[c];
  }
}

// final_state = LN(states[:, -1, :]); final_traj = global_summary[:, -1, :]
__global__ void finals_kernel(const float* __restrict__ states,
                              const float* __restrict__ gs,
                              const float* __restrict__ g4,
                              const float* __restrict__ b4,
                              float* __restrict__ out)
{
  __shared__ float red[256];
  int b = blockIdx.x, tid = threadIdx.x;
  size_t row = (size_t)(b * Tlen + (Tlen - 1)) * Dm;
  float v = states[row + tid];
  red[tid] = v; __syncthreads();
  for (int s = 128; s > 0; s >>= 1) { if (tid < s) red[tid] += red[tid + s]; __syncthreads(); }
  float mean = red[0] * (1.0f / 256.0f);
  __syncthreads();
  float dv = v - mean;
  red[tid] = dv * dv; __syncthreads();
  for (int s = 128; s > 0; s >>= 1) { if (tid < s) red[tid] += red[tid + s]; __syncthreads(); }
  float rs = rsqrtf(red[0] * (1.0f / 256.0f) + 1e-5f);
  out[(size_t)NTOK * Dm + b * Dm + tid] = dv * rs * g4[tid] + b4[tid];
  out[(size_t)NTOK * Dm + Bb * Dm + b * Dm + tid] = gs[row + tid];
}

// ---------------------------------------------------------------------------
extern "C" void kernel_launch(void* const* d_in, const int* in_sizes, int n_in,
                              void* d_out, int out_size, void* d_ws, size_t ws_size,
                              hipStream_t stream)
{
  (void)in_sizes; (void)n_in; (void)out_size; (void)ws_size;
  const float* x      = (const float*)d_in[0];
  const float* seqst  = (const float*)d_in[1];
  const float* conv_w = (const float*)d_in[3];
  const float* conv_b = (const float*)d_in[4];
  const float* qkv_w  = (const float*)d_in[5];
  const float* qkv_b  = (const float*)d_in[6];
  const float* mo_w   = (const float*)d_in[7];
  const float* mo_b   = (const float*)d_in[8];
  const float* fg_w   = (const float*)d_in[9];
  const float* fg_b   = (const float*)d_in[10];
  const float* gkv_w  = (const float*)d_in[11];
  const float* gkv_b  = (const float*)d_in[12];
  const float* go_w   = (const float*)d_in[13];
  const float* go_b   = (const float*)d_in[14];
  const float* comb_w = (const float*)d_in[15];
  const float* comb_b = (const float*)d_in[16];
  const float* ffn1_w = (const float*)d_in[17];
  const float* ffn1_b = (const float*)d_in[18];
  const float* ffn2_w = (const float*)d_in[19];
  const float* ffn2_b = (const float*)d_in[20];
  const float* tr1_w  = (const float*)d_in[21];
  const float* tr1_b  = (const float*)d_in[22];
  const float* tr2_w  = (const float*)d_in[23];
  const float* tr2_b  = (const float*)d_in[24];
  const float* p2s_w  = (const float*)d_in[25];
  const float* p2s_b  = (const float*)d_in[26];
  const float* s2p_w  = (const float*)d_in[27];
  const float* s2p_b  = (const float*)d_in[28];
  const float* ln1_g  = (const float*)d_in[29];
  const float* ln1_b  = (const float*)d_in[30];
  const float* ln2_g  = (const float*)d_in[31];
  const float* ln2_b  = (const float*)d_in[32];
  const float* ln3_g  = (const float*)d_in[33];
  const float* ln3_b  = (const float*)d_in[34];
  const float* ln4_g  = (const float*)d_in[35];
  const float* ln4_b  = (const float*)d_in[36];
  float* outp = (float*)d_out;

  float* ws = (float*)d_ws;
  const size_t SL = (size_t)NTOK * Dm;
  auto slot = [&](int i) { return ws + SL * (size_t)i; };
  float* qkvb   = slot(0);   // slots 0..2 (N x 768)
  float* lh     = slot(3);
  float* attout = slot(4);
  float* med    = slot(5);
  float* fgb    = slot(6);
  float* gkvb   = slot(7);   // slots 7..8 (N x 512)
  float* conc   = slot(9);
  float* gout   = slot(10);
  float* comb   = slot(11);
  float* pr     = slot(12);
  float* gs     = slot(13);
  // lifetime reuse
  float* cs1    = fgb;
  float* h1     = gkvb;
  float* scaled = attout;
  float* preln  = conc;
  float* states = med;
  float* ffnin  = gout;
  float* h2     = slot(0);   // N x 1024, slots 0..3 (qkv + lh dead)
  float* ffnout = comb;

  // bf16 weight arena in slot 14
  unsigned short* wbf = (unsigned short*)slot(14);
  const int SZ_QKV = 768 * 256, SZ_MO = 256 * 256, SZ_FG = 256 * 512,
            SZ_GKV = 512 * 256, SZ_GO = 256 * 256, SZ_COMB = 256 * 768,
            SZ_FFN1 = 1024 * 256, SZ_FFN2 = 256 * 1024, SZ_TR1 = 512 * 512,
            SZ_TR2 = 256 * 512, SZ_P2S = 256 * 256, SZ_S2P = 256 * 256;
  unsigned short* w_qkv  = wbf;
  unsigned short* w_mo   = w_qkv  + SZ_QKV;
  unsigned short* w_fg   = w_mo   + SZ_MO;
  unsigned short* w_gkv  = w_fg   + SZ_FG;
  unsigned short* w_go   = w_gkv  + SZ_GKV;
  unsigned short* w_comb = w_go   + SZ_GO;
  unsigned short* w_ffn1 = w_comb + SZ_COMB;
  unsigned short* w_ffn2 = w_ffn1 + SZ_FFN1;
  unsigned short* w_tr1  = w_ffn2 + SZ_FFN2;
  unsigned short* w_tr2  = w_tr1  + SZ_TR1;
  unsigned short* w_p2s  = w_tr2  + SZ_TR2;
  unsigned short* w_s2p  = w_p2s  + SZ_P2S;

  dim3 blk(256);
  auto conv = [&](const float* src, unsigned short* dst, int n) {
    tobf16_kernel<<<(n + 255) / 256, blk, 0, stream>>>(src, dst, n);
  };
  conv(qkv_w, w_qkv, SZ_QKV);   conv(mo_w, w_mo, SZ_MO);
  conv(fg_w, w_fg, SZ_FG);      conv(gkv_w, w_gkv, SZ_GKV);
  conv(go_w, w_go, SZ_GO);      conv(comb_w, w_comb, SZ_COMB);
  conv(ffn1_w, w_ffn1, SZ_FFN1);conv(ffn2_w, w_ffn2, SZ_FFN2);
  conv(tr1_w, w_tr1, SZ_TR1);   conv(tr2_w, w_tr2, SZ_TR2);
  conv(p2s_w, w_p2s, SZ_P2S);   conv(s2p_w, w_s2p, SZ_S2P);

  auto gemm = [&](const float* A, int lda, const unsigned short* W, int ldw,
                  int wOff, const float* bias, float* C, int ldc,
                  const float* resid, int K, int Nout, int accum, int actc,
                  float oscale) {
    dim3 grid(NTOK / 128, Nout / 64);
    gemm_bf16_kernel<<<grid, blk, 0, stream>>>(A, lda, W, ldw, wOff, bias,
                                               C, ldc, resid, K, accum, actc, oscale);
  };

  // 1. local history (dw conv, KLOC=8)
  dwconv8_kernel<<<NTOK * Dm / 256, blk, 0, stream>>>(x, conv_w, conv_b, lh);
  // 2. qkv projection (D -> 3D)
  gemm(x, Dm, w_qkv, Dm, 0, qkv_b, qkvb, 768, nullptr, Dm, 768, 0, 0, 1.0f);
  // 3. banded attention
  attn_kernel<<<dim3(Tlen / 16, 4, Bb), dim3(32), 0, stream>>>(qkvb, attout);
  // 4. medium output projection
  gemm(attout, Dm, w_mo, Dm, 0, mo_b, med, Dm, nullptr, Dm, Dm, 0, 0, 1.0f);
  // 5. forget gate: sigmoid([x, med] @ fg_w^T + b)
  gemm(x,   Dm, w_fg, 512, 0,   nullptr, fgb, Dm, nullptr, Dm, Dm, 0, 0, 1.0f);
  gemm(med, Dm, w_fg, 512, 256, fg_b,    fgb, Dm, nullptr, Dm, Dm, 1, 1, 1.0f);
  // 6. medium_out = med * fg
  mul_kernel<<<NTOK * Dm / 256, blk, 0, stream>>>(med, fgb, med);
  // 7. gkv (D -> 2D)
  gemm(x, Dm, w_gkv, Dm, 0, gkv_b, gkvb, 512, nullptr, Dm, 512, 0, 0, 1.0f);
  // 8. normalized cumulative concepts
  concepts_kernel<<<Bb * Dm / 256, blk, 0, stream>>>(gkvb, conc);
  // 9. global output projection
  gemm(conc, Dm, w_go, Dm, 0, go_b, gout, Dm, nullptr, Dm, Dm, 0, 0, 1.0f);
  // 10. combined = [lh, medium_out, global_out] @ comb_w^T + b
  gemm(lh,   Dm, w_comb, 768, 0,   nullptr, comb, Dm, nullptr, Dm, Dm, 0, 0, 1.0f);
  gemm(med,  Dm, w_comb, 768, 256, nullptr, comb, Dm, nullptr, Dm, Dm, 1, 0, 1.0f);
  gemm(gout, Dm, w_comb, 768, 512, comb_b,  comb, Dm, nullptr, Dm, Dm, 1, 0, 1.0f);
  // 11. pr = LN(x + combined)
  ln_kernel<<<NTOK / 8, blk, 0, stream>>>(comb, x, ln1_g, ln1_b, pr);
  // 12. global_summary: EMA recurrence (gain 0.1, decay 0.9)
  scan_kernel<<<Bb * Dm / 256, blk, 0, stream>>>(pr, gs, nullptr, 0.9f, 0.1f);
  // 13. cs1 = pr + 0.3 * (pr @ p2s^T + b)
  gemm(pr, Dm, w_p2s, Dm, 0, p2s_b, cs1, Dm, pr, Dm, Dm, 0, 0, 0.3f);
  // 14. h1 = gelu([cs1, gs] @ tr1^T + b)   (2D -> 2D)
  gemm(cs1, Dm, w_tr1, 512, 0,   nullptr, h1, 512, nullptr, Dm, 512, 0, 0, 1.0f);
  gemm(gs,  Dm, w_tr1, 512, 256, tr1_b,   h1, 512, nullptr, Dm, 512, 1, 2, 1.0f);
  // 15. scaled = 0.5 * tanh(h1 @ tr2^T + b)
  gemm(h1, 512, w_tr2, 512, 0, tr2_b, scaled, Dm, nullptr, 512, Dm, 0, 3, 0.5f);
  // 16. pre_ln recurrence seeded with sequential_state (decay 0.9)
  scan_kernel<<<Bb * Dm / 256, blk, 0, stream>>>(scaled, preln, seqst, 0.9f, 1.0f);
  // 17. states = LN(pre_ln)
  ln_kernel<<<NTOK / 8, blk, 0, stream>>>(preln, nullptr, ln3_g, ln3_b, states);
  // 18. ffn_in = pr + 0.3 * (states @ s2p^T + b)
  gemm(states, Dm, w_s2p, Dm, 0, s2p_b, ffnin, Dm, pr, Dm, Dm, 0, 0, 0.3f);
  // 19. h2 = gelu(ffn_in @ ffn1^T + b)  (D -> 4D)
  gemm(ffnin, Dm, w_ffn1, Dm, 0, ffn1_b, h2, 1024, nullptr, Dm, 1024, 0, 2, 1.0f);
  // 20. ffn_out = h2 @ ffn2^T + b       (4D -> D)
  gemm(h2, 1024, w_ffn2, 1024, 0, ffn2_b, ffnout, Dm, nullptr, 1024, Dm, 0, 0, 1.0f);
  // 21. pr_out = LN(pr + ffn_out) -> d_out
  ln_kernel<<<NTOK / 8, blk, 0, stream>>>(pr, ffnout, ln2_g, ln2_b, outp);
  // 22. final_state + final_traj
  finals_kernel<<<Bb, blk, 0, stream>>>(states, gs, ln4_g, ln4_b, outp);
}
